// TttM1Module_2370821947412
// MI455X (gfx1250) — matrix operand order repaired and checked by reading, not compiled, not run on hardware
//
#include <hip/hip_runtime.h>

// ---------------- problem constants ----------------
#define Bq   4
#define Lq   2048
#define Cq   2048
#define NHq  32
#define HDq  64
#define Kq   16
#define NCHUNK (Lq / Kq)          // 128
#define LRq  1.0f
#define EPSq 1e-6f

typedef __attribute__((ext_vector_type(16))) __bf16 bf16x16;
typedef __attribute__((ext_vector_type(8)))  __bf16 bf16x8;
typedef __attribute__((ext_vector_type(4)))  __bf16 bf16x4;
typedef __attribute__((ext_vector_type(8)))  float  f32x8;
typedef __attribute__((ext_vector_type(4)))  unsigned uint32x4;
typedef __attribute__((ext_vector_type(8)))  int      int32x8;
typedef __attribute__((ext_vector_type(4)))  int      int32x4;

// f32 -> bf16 round-to-nearest-even
__device__ __forceinline__ __bf16 to_bf16(float f) {
  union { float f; unsigned u; } v; v.f = f;
  unsigned r = v.u + 0x7FFFu + ((v.u >> 16) & 1u);
  unsigned short h = (unsigned short)(r >> 16);
  union { unsigned short s; __bf16 b; } o; o.s = h;
  return o.b;
}

// A-operand fragment (16xK tile, 16-bit elems, wave32), per CDNA5 ISA layout:
// lane L holds row M = L&15; lanes 0-15 K={k0..k0+7, k0+16..k0+23}, k0=kbase;
// lanes 16-31 same with k0=kbase+8.
__device__ __forceinline__ bf16x16 frag_a(const __bf16* __restrict__ base,
                                          int stride, int mbase, int kbase, int lane) {
  int m  = mbase + (lane & 15);
  int k0 = kbase + ((lane >> 4) << 3);
  const __bf16* p = base + m * stride + k0;
  bf16x8 lo = *(const bf16x8*)(p);
  bf16x8 hi = *(const bf16x8*)(p + 16);
  return __builtin_shufflevector(lo, hi, 0,1,2,3,4,5,6,7,8,9,10,11,12,13,14,15);
}

// B-operand fragment (Kx16 tile). Buffer holds B transposed (N-major, K contiguous):
// lane L reads column N = nbase+(L&15); lanes 0-15 K={k0..k0+7, k0+16..k0+23}, k0=kbase;
// lanes 16-31 same with k0=kbase+8.
__device__ __forceinline__ bf16x16 frag_b(const __bf16* __restrict__ baseT,
                                          int stride, int nbase, int kbase, int lane) {
  int n  = nbase + (lane & 15);
  int k0 = kbase + ((lane >> 4) << 3);
  const __bf16* p = baseT + n * stride + k0;
  bf16x8 lo = *(const bf16x8*)(p);
  bf16x8 hi = *(const bf16x8*)(p + 16);
  return __builtin_shufflevector(lo, hi, 0,1,2,3,4,5,6,7,8,9,10,11,12,13,14,15);
}

__device__ __forceinline__ f32x8 wmma_bf16(bf16x16 a, bf16x16 b, f32x8 c) {
  return __builtin_amdgcn_wmma_f32_16x16x32_bf16(false, a, false, b, (short)0, c, false, false);
}

// ---------------------------------------------------------------------------
// TDM: DMA a 2D bf16 tile (tile_d0 contiguous elems x tile_d1 rows) from global
// memory into LDS at byte offset lds_off. D# packed per CDNA5 ISA 8.3/8.4.
// This toolchain exposes the 6-arg builtin:
//   (uint32x4 g0, int32x8 g1, int32x4 g2, int32x4 g3, int32x8 extra, i32 cpol)
// ---------------------------------------------------------------------------
__device__ __forceinline__ void tdm_load_2d(unsigned lds_off, const __bf16* gptr,
                                            unsigned tensor_d0, unsigned tensor_d1,
                                            unsigned tile_d0, unsigned tile_d1,
                                            unsigned long long d0_stride) {
  unsigned long long ga = (unsigned long long)gptr;
  uint32x4 g0;
  g0[0] = 1u;                                              // count=1, user descriptor
  g0[1] = lds_off;                                         // lds_addr (bytes)
  g0[2] = (unsigned)(ga & 0xFFFFFFFFu);                    // global_addr[31:0]
  g0[3] = (unsigned)((ga >> 32) & 0x01FFFFFFu) | (2u << 30); // addr[56:32] | type=2
  int32x8 g1;
  g1[0] = (int)(1u << 16);                                 // data_size=1 -> 2 bytes
  g1[1] = (int)((tensor_d0 & 0xFFFFu) << 16);              // tensor_dim0[15:0]
  g1[2] = (int)((tensor_d0 >> 16) | ((tensor_d1 & 0xFFFFu) << 16));
  g1[3] = (int)((tensor_d1 >> 16) | (tile_d0 << 16));      // tile_dim0
  g1[4] = (int)(tile_d1 & 0xFFFFu);                        // tile_dim1 (tile_dim2=0)
  g1[5] = (int)(d0_stride & 0xFFFFFFFFu);                  // tensor_dim0_stride[31:0]
  g1[6] = (int)((d0_stride >> 32) & 0xFFFFu);              // stride[47:32]
  g1[7] = 0;
  int32x4 z4; z4[0] = 0; z4[1] = 0; z4[2] = 0; z4[3] = 0;
  int32x8 z8;
  for (int i = 0; i < 8; ++i) z8[i] = 0;
  __builtin_amdgcn_tensor_load_to_lds(g0, g1, z4, z4, z8, 0);
}

// ---------------------------------------------------------------------------
// C(f32) = A(bf16, MxK row-major) @ B (given as BT, bf16 NxK row-major).
// 256 threads = 8 waves (4x2), each wave owns a 32x64 sub-tile.
// Tiles are DMA'd by the TDM (wave 0 issues; TENSORcnt + barriers sync),
// double-buffered in 32KB dynamic LDS.
// ---------------------------------------------------------------------------
__global__ __launch_bounds__(256)
void gemm_tdm(const __bf16* __restrict__ A, const __bf16* __restrict__ BT,
              float* __restrict__ Cm, int M, int N, int Kd) {
  extern __shared__ __align__(32) char smem[];
  // buffers: A tiles (128x32 bf16 = 8KB) at 0/8192, BT tiles at 16384/24576
  const int tid  = threadIdx.x;
  const int lane = tid & 31;
  const int wave = tid >> 5;
  const int wrow = wave >> 1;     // 0..3 -> 32-row strip
  const int wcol = wave & 1;      // 0..1 -> 64-col strip
  const int bm = blockIdx.y * 128;
  const int bn = blockIdx.x * 128;

  f32x8 acc[2][4];
  for (int i = 0; i < 2; ++i)
    for (int j = 0; j < 4; ++j)
      for (int r = 0; r < 8; ++r) acc[i][j][r] = 0.0f;

  const int NK = Kd / 32;
  auto issue = [&](int kt, int sel) {
    tdm_load_2d(sel ? 8192u : 0u,      A  + (size_t)bm * Kd + kt * 32,
                (unsigned)Kd, (unsigned)M, 32u, 128u, (unsigned long long)Kd);
    tdm_load_2d(sel ? 24576u : 16384u, BT + (size_t)bn * Kd + kt * 32,
                (unsigned)Kd, (unsigned)N, 32u, 128u, (unsigned long long)Kd);
  };
  if (wave == 0) issue(0, 0);

  for (int kt = 0; kt < NK; ++kt) {
    if (wave == 0) {
      if (kt + 1 < NK) issue(kt + 1, (kt + 1) & 1);
      // 2 outstanding per tile-pair: wait until current pair has landed
      if (kt + 1 < NK) __builtin_amdgcn_s_wait_tensorcnt(2);
      else             __builtin_amdgcn_s_wait_tensorcnt(0);
    }
    __syncthreads();                       // tile kt visible to all waves
    const __bf16* sA  = (const __bf16*)(smem + ((kt & 1) ? 8192 : 0));
    const __bf16* sBT = (const __bf16*)(smem + ((kt & 1) ? 24576 : 16384));
    bf16x16 af[2], bfr[4];
    for (int i = 0; i < 2; ++i) af[i]  = frag_a(sA,  32, wrow * 32 + 16 * i, 0, lane);
    for (int j = 0; j < 4; ++j) bfr[j] = frag_b(sBT, 32, wcol * 64 + 16 * j, 0, lane);
    for (int i = 0; i < 2; ++i)
      for (int j = 0; j < 4; ++j)
        acc[i][j] = wmma_bf16(af[i], bfr[j], acc[i][j]);
    __syncthreads();                       // all reads done before buffer reuse
  }

  const int nloc = lane & 15;
  const int mhi  = (lane >> 4) << 3;
  for (int i = 0; i < 2; ++i)
    for (int j = 0; j < 4; ++j)
      for (int r = 0; r < 8; ++r) {
        int m = bm + wrow * 32 + 16 * i + mhi + r;
        int n = bn + wcol * 64 + 16 * j + nloc;
        Cm[(size_t)m * N + n] = acc[i][j][r];
      }
}

// ---------------------------------------------------------------------------
// f32 -> bf16 converters (row-major, and transposed for weight matrices)
// ---------------------------------------------------------------------------
__global__ __launch_bounds__(256)
void cvt_bf16_kernel(const float* __restrict__ src, __bf16* __restrict__ dst, int n) {
  int i = (blockIdx.x * blockDim.x + threadIdx.x) * 4;
  if (i >= n) return;
  float4 v = *(const float4*)(src + i);
  bf16x4 o;
  o[0] = to_bf16(v.x); o[1] = to_bf16(v.y); o[2] = to_bf16(v.z); o[3] = to_bf16(v.w);
  *(bf16x4*)(dst + i) = o;
}

__global__ __launch_bounds__(256)
void cvt_bf16_T_kernel(const float* __restrict__ src, __bf16* __restrict__ dstT,
                       int R, int N) {           // src RxN -> dstT NxR
  int idx = blockIdx.x * blockDim.x + threadIdx.x;
  if (idx >= R * N) return;
  int n = idx % N, r = idx / N;
  dstT[(size_t)n * R + r] = to_bf16(src[idx]);
}

// ---------------------------------------------------------------------------
// coeff[b,h,chunk,t] = LR * (1/(t+1)) * sigmoid(hs_row . ilr_w[:,h] + ilr_b[h]) / HD
// ---------------------------------------------------------------------------
__global__ __launch_bounds__(256)
void coeff_kernel(const float* __restrict__ hs, const float* __restrict__ ilr_w,
                  const float* __restrict__ ilr_b, float* __restrict__ coeff) {
  int idx = blockIdx.x * blockDim.x + threadIdx.x;    // B*L*NH
  int h = idx % NHq;
  int row = idx / NHq;
  if (row >= Bq * Lq) return;
  const float* x = hs + (size_t)row * Cq;
  float a = ilr_b[h];
  for (int k = 0; k < Cq; ++k) a += x[k] * ilr_w[k * NHq + h];
  float sig = 1.0f / (1.0f + expf(-a));
  int b = row / Lq, l = row % Lq, c = l / Kq, t = l % Kq;
  float co = (LRq / (float)(t + 1)) * sig / (float)HDq;
  coeff[(((size_t)(b * NHq + h) * NCHUNK) + c) * Kq + t] = co;
}

// ---------------------------------------------------------------------------
// TTT inner scan: one block per (b,h). 128 threads = 4 waves.
// State: W1 (64x64 f32 + bf16 transposed copy for WMMA B-operand), b1 (64 f32).
// Output XCW written directly in bf16 for the final WMMA GEMM.
// ---------------------------------------------------------------------------
__global__ __launch_bounds__(128)
void ttt_scan_kernel(const float* __restrict__ XC, const float* __restrict__ XB,
                     const float* __restrict__ XA, const float* __restrict__ coeff,
                     const float* __restrict__ W1_in, const float* __restrict__ b1_in,
                     const float* __restrict__ lnw_in, const float* __restrict__ lnb_in,
                     __bf16* __restrict__ XCW) {
  __shared__ __attribute__((aligned(32))) float  W1f[64 * 64];   // state f32
  __shared__ __attribute__((aligned(32))) __bf16 W1T[64 * 64];   // state, N-major bf16
  __shared__ float b1v[64], lnw[64], lnb[64], coS[16];
  __shared__ __attribute__((aligned(32))) __bf16 xcA[16 * 64];   // A-op, row-major
  __shared__ __attribute__((aligned(32))) __bf16 xbA[16 * 64];   // A-op + B-op of xc@xb^T
  __shared__ __attribute__((aligned(32))) __bf16 xbT[64 * 32];   // A-op: -co15*xb^T, K pad->32
  __shared__ __attribute__((aligned(32))) __bf16 gradT[64 * 32]; // B-op: grad, N-major, K pad
  __shared__ __attribute__((aligned(32))) __bf16 attnA[16 * 32]; // A-op: -co*mask*(xc@xb^T)
  __shared__ float xcF[16 * 64];   // xc f32 for residual
  __shared__ float tgt[16 * 64];   // xa - xb
  __shared__ float Z1 [16 * 64];   // Z1, later reused as Z1_bar
  __shared__ float gradF[16 * 64];
  __shared__ float b1bar[16 * 64];

  const int tid  = threadIdx.x;
  const int lane = tid & 31;
  const int wave = tid >> 5;
  const int bh = blockIdx.x;
  const int b  = bh / NHq;
  const int h  = bh % NHq;

  const float* w1g = W1_in + (size_t)h * 64 * 64;
  for (int e = tid; e < 4096; e += 128) {
    float v = w1g[e];
    W1f[e] = v;
    int m = e >> 6, n = e & 63;
    W1T[n * 64 + m] = to_bf16(v);
  }
  if (tid < 64) {
    b1v[tid] = b1_in[(size_t)h * 64 + tid];
    lnw[tid] = lnw_in[(size_t)h * 64 + tid];
    lnb[tid] = lnb_in[(size_t)h * 64 + tid];
  }
  for (int e = tid; e < 64 * 16; e += 128) {           // K = 16..31 pads
    int d = e >> 4, k = 16 + (e & 15);
    xbT[d * 32 + k]   = to_bf16(0.0f);
    gradT[d * 32 + k] = to_bf16(0.0f);
  }
  for (int e = tid; e < 16 * 16; e += 128) {
    int m = e >> 4, k = 16 + (e & 15);
    attnA[m * 32 + k] = to_bf16(0.0f);
  }
  __syncthreads();

  const int nloc = lane & 15;
  const int mhi  = (lane >> 4) << 3;

  for (int c = 0; c < NCHUNK; ++c) {
    if (tid < 16) coS[tid] = coeff[((size_t)bh * NCHUNK + c) * Kq + tid];
    __syncthreads();
    const float co15 = coS[15];

    // stage xc, xb, xa-xb (16x64)
    const size_t rowbase = (size_t)b * Lq + (size_t)c * Kq;
    for (int e = tid; e < 1024; e += 128) {
      int t = e >> 6, d = e & 63;
      size_t gi = (rowbase + t) * (NHq * HDq) + (size_t)h * HDq + d;
      float vc = XC[gi], vb = XB[gi], va = XA[gi];
      xcF[e] = vc;
      xcA[e] = to_bf16(vc);
      xbA[e] = to_bf16(vb);
      tgt[e] = va - vb;
      xbT[d * 32 + t] = to_bf16(-co15 * vb);   // -co[-1] * xb^T  (W1 update)
    }
    __syncthreads();

    // Z1 = xb @ W1 + b1
    {
      int nb = wave * 16;
      f32x8 acc; for (int r = 0; r < 8; ++r) acc[r] = 0.0f;
      acc = wmma_bf16(frag_a(xbA, 64, 0, 0,  lane), frag_b(W1T, 64, nb, 0,  lane), acc);
      acc = wmma_bf16(frag_a(xbA, 64, 0, 32, lane), frag_b(W1T, 64, nb, 32, lane), acc);
      int n = nb + nloc;
      for (int r = 0; r < 8; ++r) Z1[(mhi + r) * 64 + n] = acc[r] + b1v[n];
    }
    __syncthreads();

    // grad = ln_fused_l2_bwd(Z1, tgt) on threads 0..15 | attnA on wave 3
    if (tid < 16) {
      const int t = tid;
      const float* z = &Z1[t * 64];
      float mu = 0.0f;
      for (int d = 0; d < 64; ++d) mu += z[d];
      mu *= (1.0f / 64.0f);
      float var = 0.0f;
      for (int d = 0; d < 64; ++d) { float dv = z[d] - mu; var += dv * dv; }
      var *= (1.0f / 64.0f);
      float rstd = rsqrtf(var + EPSq);
      float sg = 0.0f, sgx = 0.0f;
      for (int d = 0; d < 64; ++d) {
        float xh = (z[d] - mu) * rstd;
        float g  = (lnw[d] * xh + lnb[d] - tgt[t * 64 + d]) * lnw[d];
        sg += g; sgx += g * xh;
      }
      float sc = rstd * (1.0f / 64.0f);
      for (int d = 0; d < 64; ++d) {
        float xh = (z[d] - mu) * rstd;
        float g  = (lnw[d] * xh + lnb[d] - tgt[t * 64 + d]) * lnw[d];
        float gr = (64.0f * g - sg - xh * sgx) * sc;
        gradF[t * 64 + d] = gr;
        gradT[d * 32 + t] = to_bf16(gr);
      }
    } else if (wave == 3) {
      f32x8 acc; for (int r = 0; r < 8; ++r) acc[r] = 0.0f;
      acc = wmma_bf16(frag_a(xcA, 64, 0, 0,  lane), frag_b(xbA, 64, 0, 0,  lane), acc);
      acc = wmma_bf16(frag_a(xcA, 64, 0, 32, lane), frag_b(xbA, 64, 0, 32, lane), acc);
      for (int r = 0; r < 8; ++r) {
        int m = mhi + r;
        float v = (nloc <= m) ? acc[r] : 0.0f;        // tril mask
        attnA[m * 32 + nloc] = to_bf16(-coS[m] * v);  // fold -co row scale
      }
    }
    __syncthreads();

    // cumulative mask@grad, b1_bar, b1 update (threads 0..63)
    if (tid < 64) {
      float run = 0.0f, bb = b1v[tid];
      for (int i = 0; i < 16; ++i) {
        run += gradF[i * 64 + tid];
        b1bar[i * 64 + tid] = bb - coS[i] * run;
      }
      b1v[tid] = b1bar[15 * 64 + tid];
    }
    __syncthreads();

    // Z1_bar = xc@W1 + attnA@grad + b1_bar (reuse Z1)
    {
      int nb = wave * 16;
      f32x8 acc; for (int r = 0; r < 8; ++r) acc[r] = 0.0f;
      acc = wmma_bf16(frag_a(attnA, 32, 0, 0, lane), frag_b(gradT, 32, nb, 0, lane), acc);
      acc = wmma_bf16(frag_a(xcA, 64, 0, 0,  lane), frag_b(W1T, 64, nb, 0,  lane), acc);
      acc = wmma_bf16(frag_a(xcA, 64, 0, 32, lane), frag_b(W1T, 64, nb, 32, lane), acc);
      int n = nb + nloc;
      for (int r = 0; r < 8; ++r) {
        int m = mhi + r;
        Z1[m * 64 + n] = acc[r] + b1bar[m * 64 + n];
      }
    }
    __syncthreads();

    // out = xc + ln_fwd(Z1_bar)  -> bf16 XCW
    if (tid < 16) {
      const int t = tid;
      const float* z = &Z1[t * 64];
      float mu = 0.0f;
      for (int d = 0; d < 64; ++d) mu += z[d];
      mu *= (1.0f / 64.0f);
      float var = 0.0f;
      for (int d = 0; d < 64; ++d) { float dv = z[d] - mu; var += dv * dv; }
      var *= (1.0f / 64.0f);
      float rstd = rsqrtf(var + EPSq);
      size_t go = (rowbase + t) * (NHq * HDq) + (size_t)h * HDq;
      for (int d = 0; d < 64; ++d) {
        float xh = (z[d] - mu) * rstd;
        XCW[go + d] = to_bf16(xcF[t * 64 + d] + lnw[d] * xh + lnb[d]);
      }
    }
    __syncthreads();

    // W1 -= co15 * xb^T @ grad (sign folded into xbT); refresh W1T
    {
      int nb = wave * 16;
      for (int mt = 0; mt < 4; ++mt) {
        int mb = mt * 16;
        f32x8 acc;
        int n = nb + nloc;
        int mrow = mb + mhi;
        for (int r = 0; r < 8; ++r) acc[r] = W1f[(mrow + r) * 64 + n];
        acc = wmma_bf16(frag_a(xbT, 32, mb, 0, lane), frag_b(gradT, 32, nb, 0, lane), acc);
        for (int r = 0; r < 8; ++r) {
          float v = acc[r];
          W1f[(mrow + r) * 64 + n] = v;
          W1T[n * 64 + (mrow + r)] = to_bf16(v);
        }
      }
    }
    __syncthreads();
  }
}

// ---------------------------------------------------------------------------
extern "C" void kernel_launch(void* const* d_in, const int* in_sizes, int n_in,
                              void* d_out, int out_size, void* d_ws, size_t ws_size,
                              hipStream_t stream) {
  const float* hs    = (const float*)d_in[0];
  const float* Wq    = (const float*)d_in[1];
  const float* Wk    = (const float*)d_in[2];
  const float* Wv    = (const float*)d_in[3];
  const float* Wo    = (const float*)d_in[4];
  const float* ilr_w = (const float*)d_in[5];
  const float* ilr_b = (const float*)d_in[6];
  const float* lnw   = (const float*)d_in[7];
  const float* lnb   = (const float*)d_in[8];
  const float* W1    = (const float*)d_in[9];
  const float* b1    = (const float*)d_in[10];
  float* out = (float*)d_out;

  const int M = Bq * Lq;                 // 8192
  const int N = NHq * HDq;               // 2048
  const size_t mat = (size_t)M * N;      // 16.7M

  // ws layout (floats then bf16 regions)
  float*  XC    = (float*)d_ws;
  float*  XB    = XC + mat;
  float*  XA    = XB + mat;
  float*  coeff = XA + mat;                              // B*NH*NCHUNK*K
  __bf16* hsb   = (__bf16*)(coeff + (size_t)Bq * NHq * NCHUNK * Kq);
  __bf16* WqT   = hsb + mat;             // each weight: Cq*N
  __bf16* WkT   = WqT + (size_t)Cq * N;
  __bf16* WvT   = WkT + (size_t)Cq * N;
  __bf16* WoT   = WvT + (size_t)Cq * N;
  __bf16* XCWb  = WoT + (size_t)N * Cq;

  // converts
  cvt_bf16_kernel<<<(int)(mat / 4 / 256), 256, 0, stream>>>(hs, hsb, (int)mat);
  cvt_bf16_T_kernel<<<(Cq * N) / 256, 256, 0, stream>>>(Wq, WqT, Cq, N);
  cvt_bf16_T_kernel<<<(Cq * N) / 256, 256, 0, stream>>>(Wk, WkT, Cq, N);
  cvt_bf16_T_kernel<<<(Cq * N) / 256, 256, 0, stream>>>(Wv, WvT, Cq, N);
  cvt_bf16_T_kernel<<<(N * Cq) / 256, 256, 0, stream>>>(Wo, WoT, N, Cq);

  dim3 gblk(256);
  dim3 ggrid(N / 128, M / 128);          // (16, 64)
  gemm_tdm<<<ggrid, gblk, 32768, stream>>>(hsb, WqT, XC, M, N, Cq);
  gemm_tdm<<<ggrid, gblk, 32768, stream>>>(hsb, WkT, XB, M, N, Cq);
  gemm_tdm<<<ggrid, gblk, 32768, stream>>>(hsb, WvT, XA, M, N, Cq);
  coeff_kernel<<<(M * NHq) / 256, 256, 0, stream>>>(hs, ilr_w, ilr_b, coeff);
  ttt_scan_kernel<<<Bq * NHq, 128, 0, stream>>>(XC, XB, XA, coeff, W1, b1, lnw, lnb, XCWb);

  dim3 ogrid(Cq / 128, M / 128);
  gemm_tdm<<<ogrid, gblk, 32768, stream>>>(XCWb, WoT, out, M, Cq, N);
}
